// Mambablock_1022202216747
// MI455X (gfx1250) — compile-verified, matrix operand order checked by reading, not run on hardware
//
#include <hip/hip_runtime.h>

// ---- problem dims (from reference) ----
#define BATCH   4
#define SEQ     2048
#define DMODEL  768
#define DINNER  1536          // 2*DMODEL
#define DSTATE  16
#define DTRANK  48
#define DTPAD   64            // DTRANK padded to a multiple of 32 for WMMA K
#define DCONV   4
#define ROWS    (BATCH*SEQ)   // 8192

typedef __bf16 v16bf __attribute__((ext_vector_type(16)));
typedef float  v8f   __attribute__((ext_vector_type(8)));
typedef unsigned short u16;

#if __has_builtin(__builtin_amdgcn_tensor_load_to_lds) && \
    __has_builtin(__builtin_amdgcn_s_wait_tensorcnt)
#define USE_TDM 1
typedef unsigned int u32x4 __attribute__((ext_vector_type(4)));
typedef int          i32x4 __attribute__((ext_vector_type(4)));
typedef int          i32x8 __attribute__((ext_vector_type(8)));
#else
#define USE_TDM 0
#endif

// round-to-nearest-even f32 -> bf16 (bit pattern)
__device__ __forceinline__ u16 f2bf(float f) {
  unsigned int u = __float_as_uint(f);
  u += 0x7FFFu + ((u >> 16) & 1u);
  return (u16)(u >> 16);
}
__device__ __forceinline__ float silu_f(float x) {
  return x * (1.0f / (1.0f + __expf(-x)));
}

// ---------------------------------------------------------------------------
// residual add + layernorm, emits res (f32, to d_out) and u (bf16, to ws)
// one block per row of 768
// ---------------------------------------------------------------------------
__global__ __launch_bounds__(256)
void k_resln(const float* __restrict__ x, const float* __restrict__ res,
             const float* __restrict__ g, const float* __restrict__ b,
             float* __restrict__ res_out, u16* __restrict__ u_bf) {
  __shared__ float s[256];
  const int row = blockIdx.x, tid = threadIdx.x;
  const float* xr = x + (size_t)row * DMODEL;
  const float* rr = res + (size_t)row * DMODEL;
  float v[3];
#pragma unroll
  for (int i = 0; i < 3; ++i) {
    int c = tid + i * 256;
    v[i] = xr[c] + rr[c];
    res_out[(size_t)row * DMODEL + c] = v[i];
  }
  s[tid] = v[0] + v[1] + v[2];
  __syncthreads();
  for (int o = 128; o > 0; o >>= 1) { if (tid < o) s[tid] += s[tid + o]; __syncthreads(); }
  float mean = s[0] * (1.0f / DMODEL);
  __syncthreads();
  float d0 = v[0]-mean, d1 = v[1]-mean, d2 = v[2]-mean;
  s[tid] = d0*d0 + d1*d1 + d2*d2;
  __syncthreads();
  for (int o = 128; o > 0; o >>= 1) { if (tid < o) s[tid] += s[tid + o]; __syncthreads(); }
  float rstd = rsqrtf(s[0] * (1.0f / DMODEL) + 1e-5f);
#pragma unroll
  for (int i = 0; i < 3; ++i) {
    int c = tid + i * 256;
    u_bf[(size_t)row * DMODEL + c] = f2bf((v[i] - mean) * rstd * g[c] + b[c]);
  }
}

// ---------------------------------------------------------------------------
// f32 -> bf16 bulk convert (weights)
// ---------------------------------------------------------------------------
__global__ __launch_bounds__(256)
void k_cvt_bf16(const float* __restrict__ s, u16* __restrict__ d, int n) {
  int i = blockIdx.x * blockDim.x + threadIdx.x;
  if (i < n) d[i] = f2bf(s[i]);
}

// W_dt (DINNER x DTRANK) -> bf16 padded to (DINNER x DTPAD)
__global__ __launch_bounds__(256)
void k_pad_wdt(const float* __restrict__ w, u16* __restrict__ d) {
  int i = blockIdx.x * blockDim.x + threadIdx.x;
  if (i >= DINNER * DTPAD) return;
  int n = i / DTPAD, k = i % DTPAD;
  d[i] = (k < DTRANK) ? f2bf(w[n * DTRANK + k]) : (u16)0;
}

// ---------------------------------------------------------------------------
// Tiled WMMA GEMM: C[M,N] = A[M,K](bf16,row) x Bt[N,K](bf16,row)^T, f32 out.
// 256 threads = 8 waves; block tile 128x128, K-step 32; wave tile 32x64.
// A tile staged by the waves (global->LDS via VGPRs, with prefetch of the
// next K-slice); B (weight) tile streamed by the Tensor Data Mover when
// available (2D descriptor: 128 rows x 32 elems, stride K, 2-byte data),
// synchronized with s_wait_tensorcnt + workgroup barrier.
// Requires M%128==0, N%128==0, K%32==0.
// ---------------------------------------------------------------------------
__global__ __launch_bounds__(256)
void k_gemm_tiled(const u16* __restrict__ A, const u16* __restrict__ Bt,
                  float* __restrict__ C, int M, int N, int K) {
  __shared__ u16 sA[128 * 32];   // 8 KB
  __shared__ u16 sB[128 * 32];   // 8 KB
  const int tid  = threadIdx.x;
  const int lane = tid & 31, wave = tid >> 5;
  const int wm = wave & 3, wn = wave >> 2;            // 4x2 wave grid
  const int m0 = blockIdx.y * 128, n0 = blockIdx.x * 128;
  const int lr = tid >> 1, lh = tid & 1;              // loader: row 0..127, 16-elt half
  const int fr = lane & 15, fk = (lane >> 4) * 16;    // fragment row / K-half

  v8f acc[2][4];
#pragma unroll
  for (int i = 0; i < 2; ++i)
#pragma unroll
    for (int j = 0; j < 4; ++j) acc[i][j] = v8f{};

#if USE_TDM
  // Flat pointers into the LDS aperture carry the LDS byte offset in the
  // low 32 bits (ISA 10.2).
  const unsigned ldsB_off = (unsigned)(uintptr_t)(void*)sB;
#endif

  for (int k0 = 0; k0 < K; k0 += 32) {
#if USE_TDM
    if (wave == 0) {
      // D# group 0: count=1 | lds_addr | 57-bit global addr | type=2
      unsigned long long ga =
          (unsigned long long)(uintptr_t)(Bt + (size_t)n0 * K + k0);
      u32x4 g0 = {};
      g0[0] = 1u;
      g0[1] = ldsB_off;
      g0[2] = (unsigned)(ga & 0xFFFFFFFFull);
      g0[3] = ((unsigned)(ga >> 32) & 0x01FFFFFFu) | 0x80000000u;
      // D# group 1: data_size=2B; tensor_dim0=K, tensor_dim1=N,
      // tile_dim0=32, tile_dim1=128, dim0 stride=K
      i32x8 g1 = {};
      g1[0] = 1 << 16;                                      // data_size=1 (2B)
      g1[1] = (int)(((unsigned)K & 0xFFFFu) << 16);         // tensor_dim0 lo
      g1[2] = (int)(((unsigned)K >> 16) |
                    (((unsigned)N & 0xFFFFu) << 16));       // dim0 hi | dim1 lo
      g1[3] = (int)(((unsigned)N >> 16) | (32u << 16));     // dim1 hi | tile_dim0
      g1[4] = 128;                                          // tile_dim1 | tile_dim2=0
      g1[5] = K;                                            // dim0 stride lo
      g1[6] = 0;
      g1[7] = 0;
      i32x4 z4 = {};
#if defined(__clang_major__) && __clang_major__ >= 23
      i32x8 z8 = {};
      __builtin_amdgcn_tensor_load_to_lds(g0, g1, z4, z4, z8, 0);
#else
      __builtin_amdgcn_tensor_load_to_lds(g0, g1, z4, z4, 0);
#endif
    }
#endif
    // stage 128x32 bf16 A tile into LDS (32 B per thread)
    const uint4* ga_p = (const uint4*)(A + (size_t)(m0 + lr) * K + k0 + lh * 16);
    uint4 ra0 = ga_p[0], ra1 = ga_p[1];
    if (k0 + 32 < K) {  // hint next K-slice into cache (global_prefetch_b8)
      __builtin_prefetch(A + (size_t)(m0 + lr) * K + k0 + 32, 0, 1);
#if !USE_TDM
      __builtin_prefetch(Bt + (size_t)(n0 + lr) * K + k0 + 32, 0, 1);
#endif
    }
    uint4* pa = (uint4*)&sA[lr * 32 + lh * 16]; pa[0] = ra0; pa[1] = ra1;
#if !USE_TDM
    const uint4* gb_p = (const uint4*)(Bt + (size_t)(n0 + lr) * K + k0 + lh * 16);
    uint4 rb0 = gb_p[0], rb1 = gb_p[1];
    uint4* pb = (uint4*)&sB[lr * 32 + lh * 16]; pb[0] = rb0; pb[1] = rb1;
#endif
#if USE_TDM
    if (wave == 0) __builtin_amdgcn_s_wait_tensorcnt(0);  // B tile landed
#endif
    __syncthreads();

    v16bf af[2], bfr[4];
#pragma unroll
    for (int i = 0; i < 2; ++i)
      af[i] = *(const v16bf*)&sA[(wm * 32 + i * 16 + fr) * 32 + fk];
#pragma unroll
    for (int j = 0; j < 4; ++j)
      bfr[j] = *(const v16bf*)&sB[(wn * 64 + j * 16 + fr) * 32 + fk];

#pragma unroll
    for (int i = 0; i < 2; ++i)
#pragma unroll
      for (int j = 0; j < 4; ++j)
        acc[i][j] = __builtin_amdgcn_wmma_f32_16x16x32_bf16(
            false, af[i], false, bfr[j], (short)0, acc[i][j], false, false);
    __syncthreads();
  }

  // C/D layout: VGPR r -> rows {r, r+8}; lanes 0-15 / 16-31 select the half
  const int cn = lane & 15, ch = lane >> 4;
#pragma unroll
  for (int i = 0; i < 2; ++i)
#pragma unroll
    for (int j = 0; j < 4; ++j)
#pragma unroll
      for (int r = 0; r < 8; ++r) {
        int rowi = m0 + wm * 32 + i * 16 + ch * 8 + r;
        int coli = n0 + wn * 64 + j * 16 + cn;
        C[(size_t)rowi * N + coli] = acc[i][j][r];
      }
}

// ---------------------------------------------------------------------------
// One 16x16 WMMA tile per wave (for N=80 x_proj). Grid must cover exactly
// (M/16)*(N/16) waves so EXEC stays all-ones around the WMMA.
// ---------------------------------------------------------------------------
__global__ __launch_bounds__(256)
void k_gemm_wave(const u16* __restrict__ A, const u16* __restrict__ Bt,
                 float* __restrict__ C, int M, int N, int K) {
  const int gtid = blockIdx.x * blockDim.x + threadIdx.x;
  const int wid = gtid >> 5, lane = gtid & 31;
  const int ntiles = N >> 4;
  const int m0 = (wid / ntiles) << 4, n0 = (wid % ntiles) << 4;
  const int r = lane & 15, kh = (lane >> 4) * 16;
  v8f acc = v8f{};
  for (int k0 = 0; k0 < K; k0 += 32) {
    v16bf a = *(const v16bf*)(A + (size_t)(m0 + r) * K + k0 + kh);
    v16bf b = *(const v16bf*)(Bt + (size_t)(n0 + r) * K + k0 + kh);
    acc = __builtin_amdgcn_wmma_f32_16x16x32_bf16(false, a, false, b, (short)0,
                                                  acc, false, false);
  }
  const int ch = lane >> 4;
#pragma unroll
  for (int q = 0; q < 8; ++q)
    C[(size_t)(m0 + ch * 8 + q) * N + (n0 + r)] = acc[q];
}

// ---------------------------------------------------------------------------
// depthwise causal conv1d (width 4) + bias + SiLU on the xs half of xz;
// emits f32 (for scan skip path) and bf16 (for x_proj GEMM)
// ---------------------------------------------------------------------------
__global__ __launch_bounds__(256)
void k_conv_silu(const float* __restrict__ xz, const float* __restrict__ w,
                 const float* __restrict__ bias, float* __restrict__ xc,
                 u16* __restrict__ xcb) {
  int i = blockIdx.x * blockDim.x + threadIdx.x;
  if (i >= ROWS * DINNER) return;
  int c = i % DINNER;
  int l = (i / DINNER) % SEQ;
  int bb = i / (DINNER * SEQ);
  float acc = bias[c];
#pragma unroll
  for (int j = 0; j < DCONV; ++j) {
    int ll = l - (DCONV - 1) + j;
    if (ll >= 0)
      acc += w[c * DCONV + j] * xz[(size_t)(bb * SEQ + ll) * (2 * DINNER) + c];
  }
  float s = silu_f(acc);
  xc[i] = s;
  xcb[i] = f2bf(s);
}

// split x_proj output into dt (bf16, K padded to 64), B, C (f32)
__global__ __launch_bounds__(256)
void k_split_proj(const float* __restrict__ proj, u16* __restrict__ dtb,
                  float* __restrict__ Bs, float* __restrict__ Cs) {
  const int W = DTPAD + 2 * DSTATE;  // 96
  int i = blockIdx.x * blockDim.x + threadIdx.x;
  if (i >= ROWS * W) return;
  int row = i / W, c = i % W;
  const float* pr = proj + (size_t)row * (DTRANK + 2 * DSTATE);
  if (c < DTPAD) {
    dtb[(size_t)row * DTPAD + c] = (c < DTRANK) ? f2bf(pr[c]) : (u16)0;
  } else if (c < DTPAD + DSTATE) {
    Bs[(size_t)row * DSTATE + (c - DTPAD)] = pr[DTRANK + (c - DTPAD)];
  } else {
    Cs[(size_t)row * DSTATE + (c - DTPAD - DSTATE)] =
        pr[DTRANK + DSTATE + (c - DTPAD - DSTATE)];
  }
}

// dt = softplus(dt_raw + b_dt), in place
__global__ __launch_bounds__(256)
void k_softplus(float* __restrict__ dt, const float* __restrict__ bdt) {
  int i = blockIdx.x * blockDim.x + threadIdx.x;
  if (i >= ROWS * DINNER) return;
  float x = dt[i] + bdt[i % DINNER];
  dt[i] = (x > 20.0f) ? x : log1pf(__expf(x));
}

// ---------------------------------------------------------------------------
// selective scan: one thread per (batch, channel); 16-wide state in VGPRs
// ---------------------------------------------------------------------------
__global__ __launch_bounds__(256)
void k_scan(const float* __restrict__ dt, const float* __restrict__ xc,
            const float* __restrict__ Bs, const float* __restrict__ Cs,
            const float* __restrict__ Alog, float* __restrict__ y) {
  int i = blockIdx.x * blockDim.x + threadIdx.x;
  if (i >= BATCH * DINNER) return;
  int b = i / DINNER, d = i % DINNER;
  float a[DSTATE], h[DSTATE];
#pragma unroll
  for (int n = 0; n < DSTATE; ++n) {
    a[n] = -__expf(Alog[d * DSTATE + n]);
    h[n] = 0.0f;
  }
  for (int l = 0; l < SEQ; ++l) {
    size_t rrow = (size_t)(b * SEQ + l);
    float dtv = dt[rrow * DINNER + d];
    float xv  = xc[rrow * DINNER + d];
    float dx  = dtv * xv;
    const float* Bp = Bs + rrow * DSTATE;
    const float* Cp = Cs + rrow * DSTATE;
    float acc = 0.0f;
#pragma unroll
    for (int n = 0; n < DSTATE; ++n) {
      h[n] = __expf(dtv * a[n]) * h[n] + dx * Bp[n];
      acc += h[n] * Cp[n];
    }
    y[rrow * DINNER + d] = acc;
  }
}

// y = (y + xc*D) * silu(z), emitted as bf16 for out_proj
__global__ __launch_bounds__(256)
void k_gate(const float* __restrict__ y, const float* __restrict__ xc,
            const float* __restrict__ xz, const float* __restrict__ Dp,
            u16* __restrict__ yb) {
  int i = blockIdx.x * blockDim.x + threadIdx.x;
  if (i >= ROWS * DINNER) return;
  int d = i % DINNER;
  size_t row = (size_t)(i / DINNER);
  float z = xz[row * (2 * DINNER) + DINNER + d];
  float v = (y[i] + xc[i] * Dp[d]) * silu_f(z);
  yb[i] = f2bf(v);
}

// ---------------------------------------------------------------------------
extern "C" void kernel_launch(void* const* d_in, const int* in_sizes, int n_in,
                              void* d_out, int out_size, void* d_ws, size_t ws_size,
                              hipStream_t stream) {
  (void)in_sizes; (void)n_in; (void)out_size; (void)ws_size;
  const float* x     = (const float*)d_in[0];
  const float* res   = (const float*)d_in[1];
  const float* g     = (const float*)d_in[2];
  const float* beta  = (const float*)d_in[3];
  const float* Win   = (const float*)d_in[4];
  const float* convw = (const float*)d_in[5];
  const float* convb = (const float*)d_in[6];
  const float* Wxp   = (const float*)d_in[7];
  const float* Wdt   = (const float*)d_in[8];
  const float* bdt   = (const float*)d_in[9];
  const float* Alog  = (const float*)d_in[10];
  const float* Dp    = (const float*)d_in[11];
  const float* Wout  = (const float*)d_in[12];

  float* out_p   = (float*)d_out;                       // (B,L,DMODEL)
  float* res_out = out_p + (size_t)ROWS * DMODEL;       // (B,L,DMODEL)

  size_t off = 0;
  auto alloc = [&](size_t bytes) -> void* {
    void* r = (char*)d_ws + off;
    off += (bytes + 255) & ~(size_t)255;
    return r;
  };
  u16*   u_bf    = (u16*)  alloc((size_t)ROWS * DMODEL * 2);
  u16*   win_bf  = (u16*)  alloc((size_t)2 * DINNER * DMODEL * 2);
  u16*   wout_bf = (u16*)  alloc((size_t)DMODEL * DINNER * 2);
  u16*   wxp_bf  = (u16*)  alloc((size_t)(DTRANK + 2 * DSTATE) * DINNER * 2);
  u16*   wdt_bf  = (u16*)  alloc((size_t)DINNER * DTPAD * 2);
  float* xz      = (float*)alloc((size_t)ROWS * 2 * DINNER * 4);
  float* xc      = (float*)alloc((size_t)ROWS * DINNER * 4);
  u16*   xc_bf   = (u16*)  alloc((size_t)ROWS * DINNER * 2);
  float* proj    = (float*)alloc((size_t)ROWS * (DTRANK + 2 * DSTATE) * 4);
  u16*   dta_bf  = (u16*)  alloc((size_t)ROWS * DTPAD * 2);
  float* Bs      = (float*)alloc((size_t)ROWS * DSTATE * 4);
  float* Cs      = (float*)alloc((size_t)ROWS * DSTATE * 4);
  float* dtbuf   = (float*)alloc((size_t)ROWS * DINNER * 4);
  float* ybuf    = (float*)alloc((size_t)ROWS * DINNER * 4);
  u16*   y_bf    = (u16*)  alloc((size_t)ROWS * DINNER * 2);

  const int T = 256;
  // 1) residual + LN (writes res half of d_out, u in bf16)
  k_resln<<<ROWS, T, 0, stream>>>(x, res, g, beta, res_out, u_bf);
  // 2) weight conversions
  k_cvt_bf16<<<(2 * DINNER * DMODEL + T - 1) / T, T, 0, stream>>>(Win, win_bf, 2 * DINNER * DMODEL);
  k_cvt_bf16<<<(DMODEL * DINNER + T - 1) / T, T, 0, stream>>>(Wout, wout_bf, DMODEL * DINNER);
  k_cvt_bf16<<<((DTRANK + 2 * DSTATE) * DINNER + T - 1) / T, T, 0, stream>>>(
      Wxp, wxp_bf, (DTRANK + 2 * DSTATE) * DINNER);
  k_pad_wdt<<<(DINNER * DTPAD + T - 1) / T, T, 0, stream>>>(Wdt, wdt_bf);
  // 3) in_proj: xz = u @ W_in^T  (8192 x 3072, K=768)
  k_gemm_tiled<<<dim3((2 * DINNER) / 128, ROWS / 128), T, 0, stream>>>(
      u_bf, win_bf, xz, ROWS, 2 * DINNER, DMODEL);
  // 4) causal depthwise conv + SiLU
  k_conv_silu<<<(ROWS * DINNER + T - 1) / T, T, 0, stream>>>(xz, convw, convb, xc, xc_bf);
  // 5) x_proj: proj = xc @ W_xproj^T  (8192 x 80, K=1536) — 2560 wave tiles
  k_gemm_wave<<<(ROWS / 16) * ((DTRANK + 2 * DSTATE) / 16) / 8, T, 0, stream>>>(
      xc_bf, wxp_bf, proj, ROWS, DTRANK + 2 * DSTATE, DINNER);
  // 6) split into dt(bf16, padded K), B, C
  k_split_proj<<<(ROWS * (DTPAD + 2 * DSTATE) + T - 1) / T, T, 0, stream>>>(proj, dta_bf, Bs, Cs);
  // 7) dt_proj: dt_raw = dt @ W_dt^T  (8192 x 1536, K=64)
  k_gemm_tiled<<<dim3(DINNER / 128, ROWS / 128), T, 0, stream>>>(
      dta_bf, wdt_bf, dtbuf, ROWS, DINNER, DTPAD);
  // 8) softplus(+bias)
  k_softplus<<<(ROWS * DINNER + T - 1) / T, T, 0, stream>>>(dtbuf, bdt);
  // 9) selective scan
  k_scan<<<(BATCH * DINNER + T - 1) / T, T, 0, stream>>>(dtbuf, xc, Bs, Cs, Alog, ybuf);
  // 10) gate
  k_gate<<<(ROWS * DINNER + T - 1) / T, T, 0, stream>>>(ybuf, xc, xz, Dp, y_bf);
  // 11) out_proj: out = y @ W_out^T  (8192 x 768, K=1536) -> d_out
  k_gemm_tiled<<<dim3(DMODEL / 128, ROWS / 128), T, 0, stream>>>(
      y_bf, wout_bf, out_p, ROWS, DMODEL, DINNER);
}
